// EncBlock_32040456028326
// MI455X (gfx1250) — matrix operand order repaired and checked by reading, not compiled, not run on hardware
//
#include <hip/hip_runtime.h>
#include <math.h>

#if __has_include(<hip/amd_detail/amd_gfx1250_TDM.h>)
#define TDM_SIX_ARGS 1
#endif

// ---------------------------------------------------------------------------
// Types for WMMA fragments (gfx1250, wave32)
// ---------------------------------------------------------------------------
typedef __attribute__((ext_vector_type(16))) __bf16 v16bf;
typedef __attribute__((ext_vector_type(8)))  float  v8f;
typedef __attribute__((ext_vector_type(4)))  unsigned int u32x4;
typedef __attribute__((ext_vector_type(8)))  int i32x8;
typedef __attribute__((ext_vector_type(4)))  int i32x4;

// Fragment register block: 8 VGPRs = 16 bf16 per lane.
union Frag {
    u32x4 q[2];
    v16bf v;
};

__device__ __forceinline__ unsigned short f2bf(float f) {
    union { float f; unsigned int u; } v;
    v.f = f;
    unsigned int r = v.u + 0x7FFFu + ((v.u >> 16) & 1u);  // round-to-nearest-even
    return (unsigned short)(r >> 16);
}

__device__ __forceinline__ unsigned lds_off_of(const void* p) {
    // Generic pointer to LDS: low 32 bits are the LDS byte offset.
    return (unsigned)(unsigned long long)p;
}

// ---------------------------------------------------------------------------
// Tensor Data Mover: async 2D tile load (bf16 elements) Global -> LDS.
// rows x cols tile, row stride `stride_elems` (all in 2-byte elements).
// D# layout per CDNA5 ISA 8.3/8.4: group0 = {flags, lds_addr, global_addr,
// type=2}; group1 = {mask/data_size, tensor dims, tile dims, dim0 stride}.
// Issue from ONE wave only; completion via TENSORcnt.
// ---------------------------------------------------------------------------
__device__ __forceinline__ void tdm_load_2d(const void* gptr, unsigned lds_off,
                                            unsigned stride_elems,
                                            unsigned rows, unsigned cols) {
    unsigned long long ga = (unsigned long long)gptr;
    u32x4 g0;
    g0[0] = 1u;                                            // count=1, user mode
    g0[1] = lds_off;                                       // lds_addr (bytes)
    g0[2] = (unsigned)ga;                                  // global_addr[31:0]
    g0[3] = (unsigned)((ga >> 32) & 0x1FFFFFFu) | (2u << 30);  // addr[56:32] | type=2
    i32x8 g1;
    g1[0] = (int)(1u << 16);                               // wg_mask=0, data_size=1 (2B)
    g1[1] = (int)((cols & 0xFFFFu) << 16);                 // tensor_dim0[15:0] @ bit48
    g1[2] = (int)(((cols >> 16) & 0xFFFFu) |               // tensor_dim0[31:16]
                  ((rows & 0xFFFFu) << 16));               // tensor_dim1[15:0]
    g1[3] = (int)(((rows >> 16) & 0xFFFFu) |               // tensor_dim1[31:16]
                  ((cols & 0xFFFFu) << 16));               // tile_dim0
    g1[4] = (int)(rows & 0xFFFFu);                         // tile_dim1 (tile_dim2=0)
    g1[5] = (int)stride_elems;                             // tensor_dim0_stride[31:0]
    g1[6] = 0;                                             // stride hi / dim1_stride lo
    g1[7] = 0;
    i32x4 z4 = {0, 0, 0, 0};
#ifdef TDM_SIX_ARGS
    i32x8 z8 = {0, 0, 0, 0, 0, 0, 0, 0};
    __builtin_amdgcn_tensor_load_to_lds(g0, g1, z4, z4, z8, 0);
#else
    __builtin_amdgcn_tensor_load_to_lds(g0, g1, z4, z4, 0);
#endif
}

// ---------------------------------------------------------------------------
// Weight convert + transpose: W fp32 [K][N] -> Wt bf16 [N][K]
// ---------------------------------------------------------------------------
__global__ void cvt_wt_kernel(const float* __restrict__ W,
                              unsigned short* __restrict__ Wt,
                              int K, int N) {
    size_t idx = (size_t)blockIdx.x * 256 + threadIdx.x;
    if (idx >= (size_t)K * N) return;
    int k = (int)(idx / N);
    int n = (int)(idx % N);
    Wt[(size_t)n * K + k] = f2bf(W[idx]);
}

// ---------------------------------------------------------------------------
// LayerNorm (fp32 in) -> bf16 out. One 256-thread block per row.
// ---------------------------------------------------------------------------
__global__ __launch_bounds__(256) void ln_bf16_kernel(
    const float* __restrict__ x, const float* __restrict__ g,
    const float* __restrict__ b, unsigned short* __restrict__ out, int D) {
    int row = blockIdx.x;
    const float* xr = x + (size_t)row * D;
    float s = 0.f, s2 = 0.f;
    for (int i = threadIdx.x; i < D; i += 256) {
        float v = xr[i];
        s += v; s2 += v * v;
    }
    for (int o = 16; o > 0; o >>= 1) {
        s  += __shfl_down(s,  o, 32);
        s2 += __shfl_down(s2, o, 32);
    }
    __shared__ float red[2][8];
    int wave = threadIdx.x >> 5, lane = threadIdx.x & 31;
    if (lane == 0) { red[0][wave] = s; red[1][wave] = s2; }
    __syncthreads();
    if (threadIdx.x == 0) {
        float a = 0.f, c = 0.f;
        for (int i = 0; i < 8; i++) { a += red[0][i]; c += red[1][i]; }
        red[0][0] = a; red[1][0] = c;
    }
    __syncthreads();
    float mu   = red[0][0] / D;
    float var  = red[1][0] / D - mu * mu;
    float rstd = rsqrtf(var + 1e-5f);
    for (int i = threadIdx.x; i < D; i += 256) {
        float v = (xr[i] - mu) * rstd * g[i] + b[i];
        out[(size_t)row * D + i] = f2bf(v);
    }
}

// ---------------------------------------------------------------------------
// Generic bf16 WMMA GEMM: C[M,N] = A[M,K] @ Wt[N,K]^T (+bias) (+epilogue)
//   MODE 0: bf16 row-major out (optional bias)
//   MODE 1: bf16 V-transpose scatter out[((b*H+h)*HD+hd)*T + t]  (bias)
//   MODE 2: bf16 row-major, exact GELU (bias)
//   MODE 3: fp32 row-major, residual add (bias)
// 256 thr (8 waves), macro-tile 128x64, wave tile 32x32, BK=32.
// Tiles staged by TDM (wave 0 issues), double-buffered in LDS.
// ---------------------------------------------------------------------------
template <int MODE>
__global__ __launch_bounds__(256) void gemm_bf16(
    const unsigned short* __restrict__ A,
    const unsigned short* __restrict__ Wt,
    const float* __restrict__ bias,
    const float* __restrict__ resid,
    void* __restrict__ out,
    int M, int N, int K,
    int Tn, int Hn, int HDn) {
    __shared__ __align__(16) unsigned short As[2][128 * 32];
    __shared__ __align__(16) unsigned short Bs[2][64 * 32];

    const int tid  = threadIdx.x;
    const int wave = tid >> 5, lane = tid & 31;
    const int wm = wave >> 1, wn = wave & 1;    // 4x2 wave grid
    const int lh = lane & 15, hi = lane >> 4;   // 16-lane half select
    const int m0 = blockIdx.y * 128, n0 = blockIdx.x * 64;

    const unsigned short* Ab = A  + (size_t)m0 * K;
    const unsigned short* Wb = Wt + (size_t)n0 * K;

    v8f acc[2][2] = {};

    // Prologue: stage buffer 0.
    if (wave == 0) {
        tdm_load_2d(Ab, lds_off_of(As[0]), K, 128, 32);
        tdm_load_2d(Wb, lds_off_of(Bs[0]), K, 64, 32);
    }

    int cur = 0;
    for (int k0 = 0; k0 < K; k0 += 32) {
        __syncthreads();   // everyone done reading buffer cur^1 (prev iter)
        if (wave == 0) {
            if (k0 + 32 < K) {
                tdm_load_2d(Ab + k0 + 32, lds_off_of(As[cur ^ 1]), K, 128, 32);
                tdm_load_2d(Wb + k0 + 32, lds_off_of(Bs[cur ^ 1]), K, 64, 32);
                __builtin_amdgcn_s_wait_tensorcnt(2);  // buffer cur complete
            } else {
                __builtin_amdgcn_s_wait_tensorcnt(0);
            }
        }
        __syncthreads();   // buffer cur visible to all waves

        const unsigned short* as = As[cur];
        const unsigned short* bs = Bs[cur];
        Frag a[2], bfr[2];
        // A fragment: lanes 0-15 row m, k {0..7,16..23}; lanes 16-31 k {8..15,24..31}
        for (int i = 0; i < 2; i++) {
            int row = wm * 32 + i * 16 + lh;
            int kk  = hi ? 8 : 0;
            a[i].q[0] = *(const u32x4*)(as + row * 32 + kk);
            a[i].q[1] = *(const u32x4*)(as + row * 32 + kk + 16);
        }
        // B fragment (from Wt=[n][k]): lanes 0-15 k {0..7,16..23}; lanes 16-31 k {8..15,24..31}
        for (int j = 0; j < 2; j++) {
            int nn = wn * 32 + j * 16 + lh;
            int kk = hi ? 8 : 0;
            bfr[j].q[0] = *(const u32x4*)(bs + nn * 32 + kk);
            bfr[j].q[1] = *(const u32x4*)(bs + nn * 32 + kk + 16);
        }
        for (int i = 0; i < 2; i++)
            for (int j = 0; j < 2; j++)
                acc[i][j] = __builtin_amdgcn_wmma_f32_16x16x32_bf16(
                    false, a[i].v, false, bfr[j].v, (short)0, acc[i][j],
                    false, false);
        cur ^= 1;
    }

    // Epilogue. C layout: lane half hi, VGPR r -> (M = 8*hi + r, N = lh).
    for (int i = 0; i < 2; i++) {
        for (int j = 0; j < 2; j++) {
            int gn = n0 + wn * 32 + j * 16 + lh;
            float bv = bias ? bias[gn] : 0.f;
            for (int r = 0; r < 8; r++) {
                int gm   = m0 + wm * 32 + i * 16 + hi * 8 + r;
                float v  = acc[i][j][r] + bv;
                if (MODE == 0) {
                    ((unsigned short*)out)[(size_t)gm * N + gn] = f2bf(v);
                } else if (MODE == 1) {
                    int bb = gm / Tn, t = gm % Tn;
                    int hh = gn / HDn, hd = gn % HDn;
                    ((unsigned short*)out)[(((size_t)bb * Hn + hh) * HDn + hd) * Tn + t] = f2bf(v);
                } else if (MODE == 2) {
                    float gl = 0.5f * v * (1.f + erff(v * 0.70710678118f));
                    ((unsigned short*)out)[(size_t)gm * N + gn] = f2bf(gl);
                } else {
                    ((float*)out)[(size_t)gm * N + gn] =
                        resid[(size_t)gm * N + gn] + v;
                }
            }
        }
    }
}

// ---------------------------------------------------------------------------
// Flash-style attention, one (b, h, 64-query tile) per 128-thread block.
// Q,K bf16 row-major [B*T, D] (head slice at col h*HD); Vt bf16 [B,H,HD,T].
// K / V^T tiles staged by TDM, double-buffered.
// ---------------------------------------------------------------------------
__global__ __launch_bounds__(128) void attn_kernel(
    const unsigned short* __restrict__ Qb,
    const unsigned short* __restrict__ Kb,
    const unsigned short* __restrict__ Vt,
    unsigned short* __restrict__ Y,
    int Bn, int Tn, int Dn, int Hn, int HDn) {
    const int nqt = Tn / 64;
    const int qt = blockIdx.x % nqt;
    const int h  = (blockIdx.x / nqt) % Hn;
    const int b  = blockIdx.x / (nqt * Hn);
    const int tid = threadIdx.x, wave = tid >> 5, lane = tid & 31;
    const int lh = lane & 15, hi = lane >> 4;
    const int q0 = qt * 64;

    __shared__ __align__(16) unsigned short Ks[2][64 * 64];   // [key][hd]
    __shared__ __align__(16) unsigned short Vs[2][64 * 64];   // [hd][key]
    __shared__ __align__(16) unsigned short Ps[4][16 * 64];   // per-wave P tile

    const unsigned short* Kbase = Kb + (size_t)(b * Tn) * Dn + h * HDn;
    const unsigned short* Vbase = Vt + (size_t)(b * Hn + h) * HDn * Tn;

    // Q fragments for this wave's 16 rows, k = hd 0..31 and 32..63.
    Frag qa[2];
    {
        int row = q0 + wave * 16 + lh;
        const unsigned short* qr = Qb + (size_t)(b * Tn + row) * Dn + h * HDn;
        int kk = hi ? 8 : 0;
        qa[0].q[0] = *(const u32x4*)(qr + kk);
        qa[0].q[1] = *(const u32x4*)(qr + kk + 16);
        qa[1].q[0] = *(const u32x4*)(qr + 32 + kk);
        qa[1].q[1] = *(const u32x4*)(qr + 32 + kk + 16);
    }

    float mrun[8], lrun[8];
    v8f o[4] = {};
    for (int r = 0; r < 8; r++) { mrun[r] = -1e30f; lrun[r] = 0.f; }
    const float scale = rsqrtf((float)HDn);

    // Prologue: stage key-tile 0.
    if (wave == 0) {
        tdm_load_2d(Kbase, lds_off_of(Ks[0]), Dn, 64, 64);
        tdm_load_2d(Vbase, lds_off_of(Vs[0]), Tn, 64, 64);
    }

    int cur = 0;
    for (int kt = 0; kt < Tn; kt += 64) {
        __syncthreads();   // done reading buffer cur^1
        if (wave == 0) {
            if (kt + 64 < Tn) {
                tdm_load_2d(Kbase + (size_t)(kt + 64) * Dn,
                            lds_off_of(Ks[cur ^ 1]), Dn, 64, 64);
                tdm_load_2d(Vbase + (kt + 64),
                            lds_off_of(Vs[cur ^ 1]), Tn, 64, 64);
                __builtin_amdgcn_s_wait_tensorcnt(2);
            } else {
                __builtin_amdgcn_s_wait_tensorcnt(0);
            }
        }
        __syncthreads();

        const unsigned short* ks = Ks[cur];
        const unsigned short* vs = Vs[cur];

        // S = (Q K^T) * scale, 16x64 per wave as 4 n-tiles.
        v8f s[4];
        for (int j = 0; j < 4; j++) {
            Frag kb0, kb1;
            int nn = j * 16 + lh;
            int kk = hi ? 8 : 0;
            kb0.q[0] = *(const u32x4*)(ks + nn * 64 + kk);
            kb0.q[1] = *(const u32x4*)(ks + nn * 64 + kk + 16);
            kb1.q[0] = *(const u32x4*)(ks + nn * 64 + 32 + kk);
            kb1.q[1] = *(const u32x4*)(ks + nn * 64 + 32 + kk + 16);
            v8f z = {};
            z = __builtin_amdgcn_wmma_f32_16x16x32_bf16(false, qa[0].v, false, kb0.v, (short)0, z, false, false);
            z = __builtin_amdgcn_wmma_f32_16x16x32_bf16(false, qa[1].v, false, kb1.v, (short)0, z, false, false);
            s[j] = z * scale;
        }

        // Online softmax per row (rows live across 16-lane halves).
        float asc[8];
        for (int r = 0; r < 8; r++) {
            float mx = fmaxf(fmaxf(s[0][r], s[1][r]), fmaxf(s[2][r], s[3][r]));
            for (int d = 8; d > 0; d >>= 1) mx = fmaxf(mx, __shfl_xor(mx, d, 32));
            float mnew  = fmaxf(mrun[r], mx);
            float alpha = __expf(mrun[r] - mnew);
            float rs = 0.f;
            for (int j = 0; j < 4; j++) {
                float p = __expf(s[j][r] - mnew);
                s[j][r] = p;
                rs += p;
            }
            for (int d = 8; d > 0; d >>= 1) rs += __shfl_xor(rs, d, 32);
            lrun[r] = lrun[r] * alpha + rs;
            mrun[r] = mnew;
            asc[r]  = alpha;
        }
        for (int jj = 0; jj < 4; jj++)
            for (int r = 0; r < 8; r++) o[jj][r] *= asc[r];

        // Re-shape P (C layout) -> A layout via wave-private LDS round-trip.
        unsigned short* pw = &Ps[wave][0];
        for (int j = 0; j < 4; j++)
            for (int r = 0; r < 8; r++)
                pw[(hi * 8 + r) * 64 + j * 16 + lh] = f2bf(s[j][r]);
        asm volatile("s_wait_dscnt 0x0" ::: "memory");

        Frag pa[2];
        {
            int kk = hi ? 8 : 0;
            pa[0].q[0] = *(const u32x4*)(pw + lh * 64 + kk);
            pa[0].q[1] = *(const u32x4*)(pw + lh * 64 + kk + 16);
            pa[1].q[0] = *(const u32x4*)(pw + lh * 64 + 32 + kk);
            pa[1].q[1] = *(const u32x4*)(pw + lh * 64 + 32 + kk + 16);
        }

        // O += P @ V using Vs=[hd][key] as B^T.
        for (int jj = 0; jj < 4; jj++) {
            Frag vb0, vb1;
            int nn = jj * 16 + lh;
            int kk = hi ? 8 : 0;
            vb0.q[0] = *(const u32x4*)(vs + nn * 64 + kk);
            vb0.q[1] = *(const u32x4*)(vs + nn * 64 + kk + 16);
            vb1.q[0] = *(const u32x4*)(vs + nn * 64 + 32 + kk);
            vb1.q[1] = *(const u32x4*)(vs + nn * 64 + 32 + kk + 16);
            o[jj] = __builtin_amdgcn_wmma_f32_16x16x32_bf16(false, pa[0].v, false, vb0.v, (short)0, o[jj], false, false);
            o[jj] = __builtin_amdgcn_wmma_f32_16x16x32_bf16(false, pa[1].v, false, vb1.v, (short)0, o[jj], false, false);
        }
        cur ^= 1;
    }

    // Normalize and write Y (bf16 row-major, head slice).
    for (int jj = 0; jj < 4; jj++) {
        for (int r = 0; r < 8; r++) {
            int qrow = q0 + wave * 16 + hi * 8 + r;
            float val = o[jj][r] / lrun[r];
            int hd = jj * 16 + lh;
            Y[(size_t)(b * Tn + qrow) * Dn + h * HDn + hd] = f2bf(val);
        }
    }
}

// ---------------------------------------------------------------------------
// Host-side orchestration
// ---------------------------------------------------------------------------
extern "C" void kernel_launch(void* const* d_in, const int* in_sizes, int n_in,
                              void* d_out, int out_size, void* d_ws, size_t ws_size,
                              hipStream_t stream) {
    constexpr int Bc = 4, Tc = 2048, Dc = 768, Hc = 12, HDc = 64, DFc = 3072;
    constexpr int Mr = Bc * Tc;  // 8192 rows

    const float* x     = (const float*)d_in[0];
    const float* ln1_g = (const float*)d_in[1];
    const float* ln1_b = (const float*)d_in[2];
    const float* Wq    = (const float*)d_in[3];
    const float* bq    = (const float*)d_in[4];
    const float* Wk    = (const float*)d_in[5];
    const float* Wv    = (const float*)d_in[6];
    const float* bv    = (const float*)d_in[7];
    const float* Wo    = (const float*)d_in[8];
    const float* bo    = (const float*)d_in[9];
    const float* ln2_g = (const float*)d_in[10];
    const float* ln2_b = (const float*)d_in[11];
    const float* W1    = (const float*)d_in[12];
    const float* b1    = (const float*)d_in[13];
    const float* W2    = (const float*)d_in[14];
    const float* b2    = (const float*)d_in[15];

    // Workspace carve-up (256B aligned slabs).
    char* ws = (char*)d_ws;
    size_t off = 0;
    auto alloc = [&](size_t bytes) -> char* {
        char* p = ws + off;
        off += (bytes + 255) & ~(size_t)255;
        return p;
    };
    unsigned short* wqt = (unsigned short*)alloc((size_t)Dc * Dc * 2);
    unsigned short* wkt = (unsigned short*)alloc((size_t)Dc * Dc * 2);
    unsigned short* wvt = (unsigned short*)alloc((size_t)Dc * Dc * 2);
    unsigned short* wot = (unsigned short*)alloc((size_t)Dc * Dc * 2);
    unsigned short* w1t = (unsigned short*)alloc((size_t)Dc * DFc * 2);
    unsigned short* w2t = (unsigned short*)alloc((size_t)DFc * Dc * 2);
    unsigned short* hb  = (unsigned short*)alloc((size_t)Mr * Dc * 2);
    unsigned short* qb  = (unsigned short*)alloc((size_t)Mr * Dc * 2);
    unsigned short* kb  = (unsigned short*)alloc((size_t)Mr * Dc * 2);
    unsigned short* vtb = (unsigned short*)alloc((size_t)Mr * Dc * 2);
    unsigned short* yb  = (unsigned short*)alloc((size_t)Mr * Dc * 2);
    unsigned short* h2b = (unsigned short*)alloc((size_t)Mr * Dc * 2);
    unsigned short* gb  = (unsigned short*)alloc((size_t)Mr * DFc * 2);
    float*          x1f = (float*)alloc((size_t)Mr * Dc * 4);

    // 1. Weight transpose+convert to bf16 [N][K].
    cvt_wt_kernel<<<(Dc * Dc + 255) / 256, 256, 0, stream>>>(Wq, wqt, Dc, Dc);
    cvt_wt_kernel<<<(Dc * Dc + 255) / 256, 256, 0, stream>>>(Wk, wkt, Dc, Dc);
    cvt_wt_kernel<<<(Dc * Dc + 255) / 256, 256, 0, stream>>>(Wv, wvt, Dc, Dc);
    cvt_wt_kernel<<<(Dc * Dc + 255) / 256, 256, 0, stream>>>(Wo, wot, Dc, Dc);
    cvt_wt_kernel<<<(Dc * DFc + 255) / 256, 256, 0, stream>>>(W1, w1t, Dc, DFc);
    cvt_wt_kernel<<<(DFc * Dc + 255) / 256, 256, 0, stream>>>(W2, w2t, DFc, Dc);

    // 2. LN1
    ln_bf16_kernel<<<Mr, 256, 0, stream>>>(x, ln1_g, ln1_b, hb, Dc);

    // 3. Q, K, V projections
    dim3 g768(Dc / 64, Mr / 128);
    gemm_bf16<0><<<g768, 256, 0, stream>>>(hb, wqt, bq, nullptr, qb, Mr, Dc, Dc, Tc, Hc, HDc);
    gemm_bf16<0><<<g768, 256, 0, stream>>>(hb, wkt, nullptr, nullptr, kb, Mr, Dc, Dc, Tc, Hc, HDc);
    gemm_bf16<1><<<g768, 256, 0, stream>>>(hb, wvt, bv, nullptr, vtb, Mr, Dc, Dc, Tc, Hc, HDc);

    // 4. Attention
    attn_kernel<<<Bc * Hc * (Tc / 64), 128, 0, stream>>>(qb, kb, vtb, yb, Bc, Tc, Dc, Hc, HDc);

    // 5. Output projection + residual (fp32)
    gemm_bf16<3><<<g768, 256, 0, stream>>>(yb, wot, bo, x, x1f, Mr, Dc, Dc, Tc, Hc, HDc);

    // 6. LN2
    ln_bf16_kernel<<<Mr, 256, 0, stream>>>(x1f, ln2_g, ln2_b, h2b, Dc);

    // 7. MLP up + GELU
    dim3 g3072(DFc / 64, Mr / 128);
    gemm_bf16<2><<<g3072, 256, 0, stream>>>(h2b, w1t, b1, nullptr, gb, Mr, DFc, Dc, Tc, Hc, HDc);

    // 8. MLP down + residual -> d_out (fp32)
    gemm_bf16<3><<<g768, 256, 0, stream>>>(gb, w2t, b2, x1f, (float*)d_out, Mr, Dc, DFc, Tc, Hc, HDc);
}
